// TripCenterLoss_min_margin_2250562863872
// MI455X (gfx1250) — compile-verified, matrix operand order checked by reading, not run on hardware
//
#include <hip/hip_runtime.h>
#include <hip/hip_bf16.h>
#include <stdint.h>

typedef __attribute__((ext_vector_type(16))) _Float16 v16h;
typedef __attribute__((ext_vector_type(8)))  _Float16 v8h;
typedef __attribute__((ext_vector_type(8)))  float    v8f;

#define N_CLASSES 1000
#define FEAT      512
#define NBATCH    16384
#define ROWS_PER_BLOCK 32
#define NBLOCKS   (NBATCH / ROWS_PER_BLOCK)   // 512

// Padded LDS row stride: 1024B of data + 16B pad. 1040 % 256 == 16, so 16
// consecutive rows at the same K hit 16 distinct 16B bank groups (64 banks).
#define AROW_BYTES 1040

// ---------------------------------------------------------------------------
// Pass 1: f32 -> f16 conversion + row squared-norms (one block per row).
// ---------------------------------------------------------------------------
__global__ __launch_bounds__(256) void conv_norm_kernel(const float* __restrict__ src,
                                                        _Float16* __restrict__ dst,
                                                        float* __restrict__ norm) {
    const int row = blockIdx.x;
    const float* s = src + (size_t)row * FEAT;
    _Float16*    d = dst + (size_t)row * FEAT;
    float acc = 0.0f;
    for (int c = threadIdx.x; c < FEAT; c += 256) {
        float v = s[c];
        d[c] = (_Float16)v;
        acc += v * v;
    }
    __shared__ float red[256];
    red[threadIdx.x] = acc;
    __syncthreads();
    for (int off = 128; off > 0; off >>= 1) {
        if (threadIdx.x < off) red[threadIdx.x] += red[threadIdx.x + off];
        __syncthreads();
    }
    if (threadIdx.x == 0) norm[row] = red[0];
}

// ---------------------------------------------------------------------------
// Pass 2: main WMMA kernel.
// Block = 32 batch rows x 1024 (padded) classes; 8 waves.
// A tile staged in LDS with padded rows (bank-conflict-free, constant-offset
// ds_load_b128). Each wave: 2 row-tiles x 2 col-tiles, 4 column groups.
// Per K-chunk: 4 ds_load_b128 (A) + 4 global_load_b128 (B) -> 4 WMMAs.
// ---------------------------------------------------------------------------
__global__ __launch_bounds__(256) void trip_main_kernel(
    const _Float16* __restrict__ xh, const _Float16* __restrict__ ch,
    const float* __restrict__ xn, const float* __restrict__ cn,
    const int* __restrict__ labels, const float* __restrict__ margin,
    float* __restrict__ partial)
{
    const int tid     = threadIdx.x;
    const int lane    = tid & 31;
    const int wave    = tid >> 5;          // 0..7
    const int l15     = lane & 15;
    const int hi      = lane >> 4;         // 0 / 1 (wave half)
    const int halfOff = hi << 3;           // K-subset shift: 0 or 8 halfs
    const int rowBase = blockIdx.x * ROWS_PER_BLOCK;

    __shared__ __align__(16) unsigned char sAb[ROWS_PER_BLOCK * AROW_BYTES]; // 32.5 KB

    // --- Stage x tile into LDS (16B granules, padded row stride). ---
    for (int g = tid; g < ROWS_PER_BLOCK * (FEAT / 8); g += 256) {
        int row  = g >> 6;          // 64 granules per row
        int gcol = g & 63;
        *(v8h*)(sAb + row * AROW_BYTES + (gcol << 4)) =
            *(const v8h*)&xh[(size_t)(rowBase + row) * FEAT + (gcol << 3)];
    }
    __syncthreads();

    // Per-lane row constants: rows rt*16 + halfOff + j (accumulator VGPR j).
    float xnr[2][8]; int lab[2][8];
#pragma unroll
    for (int rt = 0; rt < 2; ++rt)
#pragma unroll
        for (int j = 0; j < 8; ++j) {
            int r = rowBase + rt * 16 + halfOff + j;
            xnr[rt][j] = xn[r];
            lab[rt][j] = labels[r];
        }

    const float INF = __builtin_inff();
    float rmin[2][8], rtrue[2][8];
#pragma unroll
    for (int rt = 0; rt < 2; ++rt)
#pragma unroll
        for (int j = 0; j < 8; ++j) { rmin[rt][j] = INF; rtrue[rt][j] = 0.0f; }

    // Per-lane A base: row-tile 0 = row l15; row-tile 1 = row l15+16
    // (constant delta 16*AROW_BYTES folds into the ds offset immediate).
    const unsigned char* aBase = sAb + l15 * AROW_BYTES + (hi << 4);

    for (int it = 0; it < 4; ++it) {
        const int col0 = (it << 8) + (wave << 5) + l15;
        const int col1 = col0 + 16;
        const int cr0  = col0 < N_CLASSES ? col0 : N_CLASSES - 1;
        const int cr1  = col1 < N_CLASSES ? col1 : N_CLASSES - 1;
        const _Float16* brow0 = ch + (size_t)cr0 * FEAT + halfOff;
        const _Float16* brow1 = ch + (size_t)cr1 * FEAT + halfOff;

        // Prefetch next column group's center rows (gfx1250 global_prefetch_b8).
        if (it < 3) {
            __builtin_prefetch(brow0 + (size_t)256 * FEAT, 0, 1);
            __builtin_prefetch(brow1 + (size_t)256 * FEAT, 0, 1);
        }

        v8f acc00 = {}, acc01 = {}, acc10 = {}, acc11 = {};
#pragma unroll 2
        for (int k0 = 0; k0 < FEAT; k0 += 32) {
            union { v16h v; v8h h[2]; } a0, a1, b0, b1;
            // A from LDS: constant-offset ds_load_b128 off one base.
            a0.h[0] = *(const v8h*)(aBase + 2 * k0);
            a0.h[1] = *(const v8h*)(aBase + 2 * k0 + 32);
            a1.h[0] = *(const v8h*)(aBase + 2 * k0 + 16 * AROW_BYTES);
            a1.h[1] = *(const v8h*)(aBase + 2 * k0 + 16 * AROW_BYTES + 32);
            // B from global (L2-resident centers).
            b0.h[0] = *(const v8h*)(brow0 + k0);
            b0.h[1] = *(const v8h*)(brow0 + k0 + 16);
            b1.h[0] = *(const v8h*)(brow1 + k0);
            b1.h[1] = *(const v8h*)(brow1 + k0 + 16);

            acc00 = __builtin_amdgcn_wmma_f32_16x16x32_f16(
                false, a0.v, false, b0.v, (short)0, acc00, false, false);
            acc01 = __builtin_amdgcn_wmma_f32_16x16x32_f16(
                false, a0.v, false, b1.v, (short)0, acc01, false, false);
            acc10 = __builtin_amdgcn_wmma_f32_16x16x32_f16(
                false, a1.v, false, b0.v, (short)0, acc10, false, false);
            acc11 = __builtin_amdgcn_wmma_f32_16x16x32_f16(
                false, a1.v, false, b1.v, (short)0, acc11, false, false);
        }

        // --- Branchless epilogue (no EXEC manipulation). ---
        // Padded cols (>= N_CLASSES) can never equal a label, and their min
        // contribution is disabled via an additive +INF constant.
        const float cn0 = cn[cr0];
        const float cn1 = cn[cr1];
        const float addInf0 = (col0 < N_CLASSES) ? 0.0f : INF;
        const float addInf1 = (col1 < N_CLASSES) ? 0.0f : INF;
#pragma unroll
        for (int rt = 0; rt < 2; ++rt) {
#pragma unroll
            for (int j = 0; j < 8; ++j) {
                float a0j = (rt == 0) ? acc00[j] : acc10[j];
                float a1j = (rt == 0) ? acc01[j] : acc11[j];
                float d0 = xnr[rt][j] + cn0 - 2.0f * a0j;
                float d1 = xnr[rt][j] + cn1 - 2.0f * a1j;
                bool  m0 = (col0 == lab[rt][j]);
                bool  m1 = (col1 == lab[rt][j]);
                rtrue[rt][j] += (m0 ? d0 : 0.0f) + (m1 ? d1 : 0.0f);
                float e0 = d0 + (m0 ? INF : addInf0);
                float e1 = d1 + (m1 ? INF : addInf1);
                rmin[rt][j] = fminf(rmin[rt][j], fminf(e0, e1));
            }
        }
    }

    // --- Reductions: 16 lanes (columns) -> waves -> block partial ---
    __shared__ float sMin[8][ROWS_PER_BLOCK];
    __shared__ float sTrue[8][ROWS_PER_BLOCK];
    __shared__ float sTerm[ROWS_PER_BLOCK];

#pragma unroll
    for (int rt = 0; rt < 2; ++rt) {
#pragma unroll
        for (int j = 0; j < 8; ++j) {
            float mn = rmin[rt][j];
            float tr = rtrue[rt][j];
#pragma unroll
            for (int mask = 1; mask < 16; mask <<= 1) {
                mn = fminf(mn, __shfl_xor(mn, mask, 16));
                tr += __shfl_xor(tr, mask, 16);
            }
            if (l15 == 0) {
                sMin[wave][rt * 16 + halfOff + j]  = mn;
                sTrue[wave][rt * 16 + halfOff + j] = tr;
            }
        }
    }
    __syncthreads();

    if (tid < ROWS_PER_BLOCK) {
        float dmin = INF, dtrue = 0.0f;
#pragma unroll
        for (int w = 0; w < 8; ++w) {
            dmin  = fminf(dmin, sMin[w][tid]);
            dtrue += sTrue[w][tid];
        }
        float mg = margin[0];
        sTerm[tid] = fmaxf(mg + dtrue - dmin, 0.0f);
    }
    __syncthreads();
    if (tid == 0) {
        float s = 0.0f;
#pragma unroll
        for (int r = 0; r < ROWS_PER_BLOCK; ++r) s += sTerm[r];
        partial[blockIdx.x] = s * (1.0f / (float)NBATCH);
    }
}

// ---------------------------------------------------------------------------
// Pass 3: deterministic scalar reduction of per-block partials.
// ---------------------------------------------------------------------------
__global__ __launch_bounds__(256) void finalize_kernel(const float* __restrict__ partial,
                                                       int n, float* __restrict__ out) {
    __shared__ float red[256];
    float acc = 0.0f;
    for (int i = threadIdx.x; i < n; i += 256) acc += partial[i];
    red[threadIdx.x] = acc;
    __syncthreads();
    for (int off = 128; off > 0; off >>= 1) {
        if (threadIdx.x < off) red[threadIdx.x] += red[threadIdx.x + off];
        __syncthreads();
    }
    if (threadIdx.x == 0) out[0] = red[0];
}

extern "C" void kernel_launch(void* const* d_in, const int* in_sizes, int n_in,
                              void* d_out, int out_size, void* d_ws, size_t ws_size,
                              hipStream_t stream) {
    const float* x       = (const float*)d_in[0];
    const int*   labels  = (const int*)d_in[1];
    const float* centers = (const float*)d_in[2];
    const float* margin  = (const float*)d_in[3];
    float*       out     = (float*)d_out;

    // Workspace layout (all offsets 256B-aligned):
    uint8_t* ws = (uint8_t*)d_ws;
    _Float16* xh      = (_Float16*)(ws + 0);          // 16384*512*2 = 16,777,216 B
    _Float16* chh     = (_Float16*)(ws + 16777216);   //  1000*512*2 =  1,024,000 B
    float*    xn      = (float*)(ws + 17801216);      // 16384*4     =     65,536 B
    float*    cnrm    = (float*)(ws + 17866752);      //  1000*4     ->      4,096 B
    float*    partial = (float*)(ws + 17870848);      //   512*4     =      2,048 B

    conv_norm_kernel<<<NBATCH,    256, 0, stream>>>(x,       xh,  xn);
    conv_norm_kernel<<<N_CLASSES, 256, 0, stream>>>(centers, chh, cnrm);
    trip_main_kernel<<<NBLOCKS,   256, 0, stream>>>(xh, chh, xn, cnrm, labels, margin, partial);
    finalize_kernel<<<1,          256, 0, stream>>>(partial, NBLOCKS, out);
}
